// SignedDynamicGNN_74002286510433
// MI455X (gfx1250) — matrix operand order repaired and checked by reading, not compiled, not run on hardware
//
#include <hip/hip_runtime.h>
#include <hip/hip_bf16.h>

#define N_NODES 20000
#define N_EDGES 1280000
#define D_FEAT  512

typedef __attribute__((ext_vector_type(16))) __bf16 v16bf;
typedef __attribute__((ext_vector_type(8)))  float  v8f;

union BFrag { v16bf v; unsigned u[8]; uint4 q[2]; };

// Native bf16 convert via fptrunc: lowers to v_cvt_*bf16_f32 on gfx1250.
__device__ __forceinline__ unsigned pack2bf(float lo, float hi) {
    union { __bf16 b[2]; unsigned u; } t;
    t.b[0] = (__bf16)lo;
    t.b[1] = (__bf16)hi;
    return t.u;
}

// ---------------- K0: init workspace (agg=0, deg=1 for self-loops) ----------
__global__ __launch_bounds__(256) void k_init(float* __restrict__ ws) {
    int t = blockIdx.x * 256 + threadIdx.x;
    if (t < 640000)       ws[640000 + t] = 0.0f;             // agg_pos, agg_neg
    else if (t < 680000)  ws[1280000 + (t - 640000)] = 1.0f; // deg_pos, deg_neg
}

// ---------------- K1: xw_pos = x@Wp, xw_neg = x@Wn via bf16 WMMA ------------
// Block = 4 waves; each wave owns one 16-row tile; both W matrices staged in
// LDS as prepacked bf16 B-fragments (fragment-order => ds_load_b128 reads).
// A fragments: two contiguous 32B regions per lane per chunk -> 4x b128 loads.
__global__ __launch_bounds__(128) void k_gemm_xw(const float* __restrict__ x,
                                                 const float* __restrict__ Wp,
                                                 const float* __restrict__ Wn,
                                                 float* __restrict__ xwp,
                                                 float* __restrict__ xwn) {
    __shared__ unsigned ldsb[2][4096]; // [mat][chunk*256 + lane*8 + i]
    const int tid = threadIdx.x;
    for (int p = tid; p < 4096; p += 128) {
        int chunk = p >> 8;
        int lane  = (p >> 3) & 31;
        int i     = p & 7;
        int hh = lane >> 4, cc = lane & 15;
        int k = chunk * 32 + 8 * hh + 2 * i + ((i >= 4) ? 8 : 0); // B layout: table order (8-K groups per half)
        ldsb[0][p] = pack2bf(Wp[k * 16 + cc], Wp[(k + 1) * 16 + cc]);
        ldsb[1][p] = pack2bf(Wn[k * 16 + cc], Wn[(k + 1) * 16 + cc]);
    }
    __syncthreads();

    const int wave = tid >> 5, lane = tid & 31;
    const int tile = blockIdx.x * 4 + wave;
    if (tile >= 1250) return;                      // 20000 = 1250*16 (no row tail)
    const int rowbase = tile * 16;
    const int h = lane >> 4, c = lane & 15;
    const float* xrow = x + (size_t)(rowbase + c) * D_FEAT;

    v8f accp = {}; v8f accn = {};
    for (int chunk = 0; chunk < 16; ++chunk) {
        const float* base = xrow + chunk * 32 + 8 * h;
        float4 r0 = *(const float4*)(base);
        float4 r1 = *(const float4*)(base + 4);
        float4 r2 = *(const float4*)(base + 16);
        float4 r3 = *(const float4*)(base + 20);
        BFrag A;
        A.u[0] = pack2bf(r0.x, r0.y); A.u[1] = pack2bf(r0.z, r0.w);
        A.u[2] = pack2bf(r1.x, r1.y); A.u[3] = pack2bf(r1.z, r1.w);
        A.u[4] = pack2bf(r2.x, r2.y); A.u[5] = pack2bf(r2.z, r2.w);
        A.u[6] = pack2bf(r3.x, r3.y); A.u[7] = pack2bf(r3.z, r3.w);

        BFrag Bp, Bn;
        const uint4* sp = (const uint4*)&ldsb[0][chunk * 256 + lane * 8];
        Bp.q[0] = sp[0]; Bp.q[1] = sp[1];
        const uint4* sn = (const uint4*)&ldsb[1][chunk * 256 + lane * 8];
        Bn.q[0] = sn[0]; Bn.q[1] = sn[1];
        accp = __builtin_amdgcn_wmma_f32_16x16x32_bf16(false, A.v, false, Bp.v,
                                                       (short)0, accp, false, false);
        accn = __builtin_amdgcn_wmma_f32_16x16x32_bf16(false, A.v, false, Bn.v,
                                                       (short)0, accn, false, false);
    }
#pragma unroll
    for (int j = 0; j < 8; ++j) {
        int row = rowbase + 8 * h + j;             // C layout: M=j (+8 for hi half)
        xwp[row * 16 + c] = accp[j];
        xwn[row * 16 + c] = accn[j];
    }
}

// ---------------- K2: degree accumulation (dst side, both graphs) -----------
__global__ __launch_bounds__(256) void k_deg(const int* __restrict__ ep,
                                             const int* __restrict__ en,
                                             float* __restrict__ degp,
                                             float* __restrict__ degn) {
    int t = blockIdx.x * 256 + threadIdx.x;        // exactly 2*E threads
    if (t < N_EDGES) atomicAdd(&degp[ep[N_EDGES + t]], 1.0f);
    else             atomicAdd(&degn[en[N_EDGES + (t - N_EDGES)]], 1.0f);
}

// ---------------- K3: deg -> rsqrt(deg) in place ----------------------------
__global__ __launch_bounds__(256) void k_dinv(float* __restrict__ d) {
    int t = blockIdx.x * 256 + threadIdx.x;
    if (t < 40000) d[t] = rsqrtf(d[t]);            // deg >= 1 always
}

// ---------------- K4: edge scatter: agg[dst] += norm * xw[src] --------------
// 16 threads per edge -> coalesced 64B gather + 64B atomic burst.
__global__ __launch_bounds__(256) void k_scatter(const int* __restrict__ ei,
                                                 const float* __restrict__ xw,
                                                 const float* __restrict__ dinv,
                                                 float* __restrict__ agg) {
    unsigned t = blockIdx.x * 256u + threadIdx.x;  // exactly E*16 threads
    unsigned e = t >> 4, f = t & 15u;
    int src = ei[e], dst = ei[N_EDGES + e];
    float norm = dinv[src] * dinv[dst];
    atomicAdd(&agg[dst * 16 + f], norm * xw[src * 16 + f]);
}

// ---------------- K5: add self-loop term + bias, ReLU, signed combine -------
__global__ __launch_bounds__(256) void k_finalize(const float* __restrict__ xwp,
                                                  const float* __restrict__ xwn,
                                                  const float* __restrict__ aggp,
                                                  const float* __restrict__ aggn,
                                                  const float* __restrict__ dinvp,
                                                  const float* __restrict__ dinvn,
                                                  const float* __restrict__ bp,
                                                  const float* __restrict__ bn,
                                                  float* __restrict__ comb) {
    int t = blockIdx.x * 256 + threadIdx.x;        // exactly 320000 threads
    int n = t >> 4, f = t & 15;
    float dp = dinvp[n], dn = dinvn[n];
    float vp = aggp[t] + dp * dp * xwp[t] + bp[f];
    float vn = aggn[t] + dn * dn * xwn[t] + bn[f];
    comb[t] = fmaxf(vp, 0.0f) - fmaxf(vn, 0.0f);
}

// ---------------- K5b: pre-pack w_ih^T (K padded 48->64) into bf16 frags ----
// wihb[((chunk*12 + nt)*32 + lane)*8 + i]; block-invariant, computed once.
__global__ __launch_bounds__(256) void k_pack_wih(const float* __restrict__ w_ih,
                                                  unsigned* __restrict__ wihb) {
    int p = blockIdx.x * 256 + threadIdx.x;        // exactly 6144 threads
    int i    = p & 7;
    int lane = (p >> 3) & 31;
    int ctn  = p >> 8;                             // chunk*12 + nt
    int chunk = ctn / 12, nt = ctn % 12;
    int h = lane >> 4, c = lane & 15;
    int k  = chunk * 32 + 8 * h + 2 * i + ((i >= 4) ? 8 : 0); // B layout: table order (8-K groups per half)
    int cg = nt * 16 + c;                          // output column 0..191
    float f0 = (k     < 48) ? w_ih[cg * 48 + k]     : 0.0f;
    float f1 = (k + 1 < 48) ? w_ih[cg * 48 + k + 1] : 0.0f;
    wihb[p] = pack2bf(f0, f1);
}

// ---------------- K6: xi = [comb | emb[round[batch]]] @ w_ih^T + b_ih -------
// A side fully vectorized (float4 from comb/emb, chunk-1 upper half is zero);
// B fragments streamed from the pre-packed L2-resident buffer via b128 loads.
__global__ __launch_bounds__(128) void k_gemm_xi(const float* __restrict__ comb,
                                                 const int* __restrict__ batch,
                                                 const int* __restrict__ rounds,
                                                 const float* __restrict__ emb,
                                                 const unsigned* __restrict__ wihb,
                                                 const float* __restrict__ b_ih,
                                                 float* __restrict__ xi) {
    const int tid  = threadIdx.x;
    const int wave = tid >> 5, lane = tid & 31;
    const int h = lane >> 4, c = lane & 15;
    const int rowbase = blockIdx.x * 16;
    const int arow = rowbase + c;
    const int ridx = rounds[batch[arow]] - 1;
    const float* crow = comb + arow * 16;
    const float* erow = emb + ridx * 32;

    float4 c0 = *(const float4*)(crow + 8 * h);
    float4 c1 = *(const float4*)(crow + 8 * h + 4);
    float4 e0 = *(const float4*)(erow + 8 * h);
    float4 e1 = *(const float4*)(erow + 8 * h + 4);
    float4 e2 = *(const float4*)(erow + 16 + 8 * h);
    float4 e3 = *(const float4*)(erow + 16 + 8 * h + 4);
    BFrag A0, A1;
    A0.u[0] = pack2bf(c0.x, c0.y); A0.u[1] = pack2bf(c0.z, c0.w);
    A0.u[2] = pack2bf(c1.x, c1.y); A0.u[3] = pack2bf(c1.z, c1.w);
    A0.u[4] = pack2bf(e0.x, e0.y); A0.u[5] = pack2bf(e0.z, e0.w);
    A0.u[6] = pack2bf(e1.x, e1.y); A0.u[7] = pack2bf(e1.z, e1.w);
    A1.u[0] = pack2bf(e2.x, e2.y); A1.u[1] = pack2bf(e2.z, e2.w);
    A1.u[2] = pack2bf(e3.x, e3.y); A1.u[3] = pack2bf(e3.z, e3.w);
    A1.u[4] = 0u; A1.u[5] = 0u; A1.u[6] = 0u; A1.u[7] = 0u;   // K pad 48->64

    v8f acc0 = {}; v8f acc1 = {}; v8f acc2 = {};
#pragma unroll
    for (int chunk = 0; chunk < 2; ++chunk) {
#pragma unroll
        for (int nt = 0; nt < 3; ++nt) {
            int nt_g = wave * 3 + nt;
            const uint4* bsrc = (const uint4*)&wihb[((chunk * 12 + nt_g) * 32 + lane) * 8];
            BFrag B; B.q[0] = bsrc[0]; B.q[1] = bsrc[1];
            const v16bf& Av = (chunk == 0) ? A0.v : A1.v;
            if (nt == 0) acc0 = __builtin_amdgcn_wmma_f32_16x16x32_bf16(false, Av, false, B.v, (short)0, acc0, false, false);
            if (nt == 1) acc1 = __builtin_amdgcn_wmma_f32_16x16x32_bf16(false, Av, false, B.v, (short)0, acc1, false, false);
            if (nt == 2) acc2 = __builtin_amdgcn_wmma_f32_16x16x32_bf16(false, Av, false, B.v, (short)0, acc2, false, false);
        }
    }
#pragma unroll
    for (int nt = 0; nt < 3; ++nt) {
        int col = wave * 48 + nt * 16 + c;
        float bias = b_ih[col];
#pragma unroll
        for (int j = 0; j < 8; ++j) {
            int row = rowbase + 8 * h + j;
            float v = (nt == 0) ? acc0[j] : (nt == 1) ? acc1[j] : acc2[j];
            xi[(size_t)row * 192 + col] = v + bias;
        }
    }
}

// ---------------- K7: sequential GRU scan + classifier + log_softmax --------
// One 64-thread block. Thread i owns hidden unit i: w_hh rows i, 64+i, 128+i
// live in REGISTERS (3x64 floats, fully unrolled constant indexing). LDS holds
// only h[64]; the inner loop is 64 LDS broadcasts + 3 independent FMA chains.
// xi is streamed with global_prefetch 8 steps ahead.
__global__ __launch_bounds__(64) void k_gru(const float* __restrict__ xi,
                                            const float* __restrict__ w_hh,
                                            const float* __restrict__ b_hh,
                                            const float* __restrict__ cls_w,
                                            const float* __restrict__ cls_b,
                                            float* __restrict__ out) {
    __shared__ float sh[64];
    const int i = threadIdx.x;

    float w0[64], w1[64], w2[64];
#pragma unroll
    for (int k = 0; k < 64; k += 4) {
        float4 a = *(const float4*)(w_hh + (size_t)i * 64 + k);
        float4 b = *(const float4*)(w_hh + (size_t)(64 + i) * 64 + k);
        float4 g = *(const float4*)(w_hh + (size_t)(128 + i) * 64 + k);
        w0[k] = a.x; w0[k + 1] = a.y; w0[k + 2] = a.z; w0[k + 3] = a.w;
        w1[k] = b.x; w1[k + 1] = b.y; w1[k + 2] = b.z; w1[k + 3] = b.w;
        w2[k] = g.x; w2[k + 1] = g.y; w2[k + 2] = g.z; w2[k + 3] = g.w;
    }
    sh[i] = 0.0f;
    const float bh0 = b_hh[i], bh1 = b_hh[64 + i], bh2 = b_hh[128 + i];
    __syncthreads();

    for (int t = 0; t < N_NODES; ++t) {
        const float* xr = xi + (size_t)t * 192;
        float x0 = xr[i], x1 = xr[64 + i], x2 = xr[128 + i];
        if (t + 8 < N_NODES)
            __builtin_prefetch(xr + 8 * 192 + i, 0, 0);   // global_prefetch_b8
        float a0 = bh0, a1 = bh1, a2 = bh2;
        float hold = sh[i];
#pragma unroll
        for (int k = 0; k < 64; ++k) {
            float hk = sh[k];
            a0 = fmaf(hk, w0[k], a0);
            a1 = fmaf(hk, w1[k], a1);
            a2 = fmaf(hk, w2[k], a2);
        }
        float r  = 1.0f / (1.0f + __expf(-(x0 + a0)));
        float z  = 1.0f / (1.0f + __expf(-(x1 + a1)));
        float ng = fminf(fmaxf(x2 + r * a2, -15.0f), 15.0f);
        float e  = __expf(2.0f * ng);
        float n  = (e - 1.0f) / (e + 1.0f);
        float hn = (1.0f - z) * n + z * hold;
        __syncthreads();
        sh[i] = hn;
        __syncthreads();
    }

    if (i == 0) {
        float l0 = cls_b[0], l1 = cls_b[1];
        for (int k = 0; k < 64; ++k) {
            l0 += sh[k] * cls_w[k * 2 + 0];
            l1 += sh[k] * cls_w[k * 2 + 1];
        }
        float m   = fmaxf(l0, l1);
        float lse = m + __logf(__expf(l0 - m) + __expf(l1 - m));
        out[0] = l0 - lse;
        out[1] = l1 - lse;
    }
}

extern "C" void kernel_launch(void* const* d_in, const int* in_sizes, int n_in,
                              void* d_out, int out_size, void* d_ws, size_t ws_size,
                              hipStream_t stream) {
    const float* x      = (const float*)d_in[0];
    const int*   ep     = (const int*)  d_in[1];
    const int*   en     = (const int*)  d_in[2];
    const int*   batch  = (const int*)  d_in[3];
    const int*   rounds = (const int*)  d_in[4];
    const float* Wp     = (const float*)d_in[5];
    const float* bp     = (const float*)d_in[6];
    const float* Wn     = (const float*)d_in[7];
    const float* bn     = (const float*)d_in[8];
    const float* emb    = (const float*)d_in[9];
    const float* w_ih   = (const float*)d_in[10];
    const float* w_hh   = (const float*)d_in[11];
    const float* b_ih   = (const float*)d_in[12];
    const float* b_hh   = (const float*)d_in[13];
    const float* cls_w  = (const float*)d_in[14];
    const float* cls_b  = (const float*)d_in[15];
    float* ws    = (float*)d_ws;
    float* xwp   = ws;                       // 320000
    float* xwn   = ws + 320000;              // 320000
    float* aggp  = ws + 640000;              // 320000 (reused as comb)
    float* aggn  = ws + 960000;              // 320000
    float* dinvp = ws + 1280000;             // 20000
    float* dinvn = ws + 1300000;             // 20000
    float* xi    = ws + 1320000;             // 3,840,000
    unsigned* wihb = (unsigned*)(ws + 5160000); // 6144 words
    float* out   = (float*)d_out;

    k_init    <<<2657, 256, 0, stream>>>(ws);
    k_pack_wih<<<24, 256, 0, stream>>>(w_ih, wihb);
    k_gemm_xw <<<313, 128, 0, stream>>>(x, Wp, Wn, xwp, xwn);
    k_deg     <<<10000, 256, 0, stream>>>(ep, en, dinvp, dinvn);
    k_dinv    <<<157, 256, 0, stream>>>(dinvp);
    k_scatter <<<80000, 256, 0, stream>>>(ep, xwp, dinvp, aggp);
    k_scatter <<<80000, 256, 0, stream>>>(en, xwn, dinvn, aggn);
    k_finalize<<<1250, 256, 0, stream>>>(xwp, xwn, aggp, aggn, dinvp, dinvn, bp, bn, aggp);
    k_gemm_xi <<<1250, 128, 0, stream>>>(aggp, batch, rounds, emb, wihb, b_ih, xi);
    k_gru     <<<1, 64, 0, stream>>>(xi, w_hh, b_hh, cls_w, cls_b, out);
}
